// GRUTrajectoryPredict_23905787969792
// MI455X (gfx1250) — matrix operand order repaired and checked by reading, not compiled, not run on hardware
//
#include <hip/hip_runtime.h>
#include <hip/hip_bf16.h>

typedef __bf16 bf16;
typedef __attribute__((ext_vector_type(16))) __bf16        v16bf;
typedef __attribute__((ext_vector_type(8)))  float         v8f;
typedef __attribute__((ext_vector_type(4)))  unsigned int  u32x4;

#define PRED_LEN 10
#define WAVES 4
#define BLOCK (WAVES * 32)
#define ROWS_PER_BLOCK (WAVES * 16)

// bf16 weight regions inside d_ws (element offsets)
#define W1_ELEMS  (256 * 384)   // K padded 361 -> 384
#define W2_ELEMS  (128 * 256)
#define W3_ELEMS  (64  * 128)
#define WHH_ELEMS (192 * 64)
#define OFF_W1  0
#define OFF_W2  (OFF_W1 + W1_ELEMS)
#define OFF_W3  (OFF_W2 + W2_ELEMS)
#define OFF_WHH (OFF_W3 + W3_ELEMS)
#define TOT_W   (OFF_WHH + WHH_ELEMS)

__device__ __forceinline__ bf16 to_bf16(float f) { return (bf16)f; }

__device__ __forceinline__ v16bf load_frag(const bf16* p0, const bf16* p1) {
  union { v16bf v; u32x4 u[2]; } f;
  f.u[0] = *(const u32x4*)p0;   // 8 bf16
  f.u[1] = *(const u32x4*)p1;   // 8 bf16
  return f.v;
}

__device__ __forceinline__ v8f wmma_bf16(v16bf a, v16bf b, v8f c) {
  return __builtin_amdgcn_wmma_f32_16x16x32_bf16(false, a, false, b, (short)0, c,
                                                 false, false);
}

// One 16-row GEMM stage: [16 x (KT*32)] @ W^T (W is [N][K] bf16, row-major).
// NT n-tiles of 16 columns, KT k-tiles of 32. Bias always added.
template <int NT, int KT, int LDI, int LDK, int LDO, bool RELU, bool BF_OUT, bool F_OUT>
__device__ __forceinline__ void gemm_stage(const bf16* __restrict__ inT,
                                           const bf16* __restrict__ W,
                                           const float* __restrict__ bias,
                                           bf16* bfOut, float* fOut, int lane) {
  const int col = lane & 15;
  const int hi  = lane >> 4;               // 0: low K half/rows, 1: high
  const bf16* aRow = inT + col * LDI;      // A-matrix: lane row = lane & 15

  for (int nt = 0; nt < NT; ++nt) {
    const int n = nt * 16 + col;
    const float bv = bias[n];
    const bf16* wRow = W + n * LDK;        // B column n == weight row n
    v8f acc = {};
    for (int kt = 0; kt < KT; ++kt) {
      const int ka = kt * 32 + hi * 8;     // A: K {0..7,16..23} / {8..15,24..31}
      v16bf a = load_frag(aRow + ka, aRow + ka + 16);
      const int kb = kt * 32 + hi * 8;     // B: K {0..7,16..23} / {8..15,24..31}
      v16bf b = load_frag(wRow + kb, wRow + kb + 16);
      acc = wmma_bf16(a, b, acc);
    }
#pragma unroll
    for (int r = 0; r < 8; ++r) {
      float v = acc[r] + bv;
      if (RELU) v = fmaxf(v, 0.0f);
      const int row = r + hi * 8;          // C/D layout
      if (BF_OUT) bfOut[row * LDO + n] = to_bf16(v);
      if (F_OUT)  fOut[row * LDO + n]  = v;
    }
  }
}

__global__ void __launch_bounds__(BLOCK) convert_weights_kernel(
    const float* __restrict__ W1, const float* __restrict__ W2,
    const float* __restrict__ W3, const float* __restrict__ Whh,
    bf16* __restrict__ ws) {
  int i = blockIdx.x * blockDim.x + threadIdx.x;
  if (i < W1_ELEMS) {                      // pad K 361 -> 384 with zeros
    int r = i / 384, c = i - r * 384;
    ws[OFF_W1 + i] = to_bf16(c < 361 ? W1[r * 361 + c] : 0.0f);
    return;
  }
  i -= W1_ELEMS;
  if (i < W2_ELEMS) { ws[OFF_W2 + i] = to_bf16(W2[i]); return; }
  i -= W2_ELEMS;
  if (i < W3_ELEMS) { ws[OFF_W3 + i] = to_bf16(W3[i]); return; }
  i -= W3_ELEMS;
  if (i < WHH_ELEMS) { ws[OFF_WHH + i] = to_bf16(Whh[i]); }
}

__global__ void __launch_bounds__(BLOCK) gru_traj_kernel(
    const float* __restrict__ z, const bf16* __restrict__ wsb,
    const float* __restrict__ b1, const float* __restrict__ b2,
    const float* __restrict__ b3, const float* __restrict__ Wih,
    const float* __restrict__ bih, const float* __restrict__ bhh,
    const float* __restrict__ Wo, const float* __restrict__ bo,
    float* __restrict__ out) {
  // per-wave private LDS regions (no cross-wave sharing)
  __shared__ __align__(16) bf16 sA[WAVES][16 * 384];  // 12288 B each
  __shared__ __align__(16) bf16 sB[WAVES][16 * 256];  //  8192 B each

  const int tid = threadIdx.x;
  const int w = tid >> 5;
  const int lane = tid & 31;
  const int rowBase = blockIdx.x * ROWS_PER_BLOCK + w * 16;

  bf16* aT = sA[w];
  bf16* bT = sB[w];

  // ---- stage 0: z tile [16 x 361] f32 -> LDS bf16 [16 x 384] (zero padded)
  const float* zBase = z + (long long)rowBase * 361;
  for (int idx = lane; idx < 16 * 384; idx += 32) {   // 192 iters, uniform
    int row = idx / 384, c = idx - row * 384;
    float v = (c < 361) ? zBase[row * 361 + c] : 0.0f;
    aT[idx] = to_bf16(v);
  }
  __syncthreads();

  // ---- MLP: 384->256->128->64, ReLU
  gemm_stage<16, 12, 384, 384, 256, true, true, false>(aT, wsb + OFF_W1, b1, bT, nullptr, lane);
  __syncthreads();
  gemm_stage<8, 8, 256, 256, 128, true, true, false>(bT, wsb + OFF_W2, b2, aT, nullptr, lane);
  __syncthreads();

  bf16*  hb = bT;                       // h as bf16 [16x64]   (2048 B)
  float* hF = (float*)(bT + 1024);      // h as f32  [16x64]   (4096 B)
  float* xb = (float*)(bT + 3072);      // x        [16x2]    (128 B)
  gemm_stage<4, 4, 128, 128, 64, true, true, true>(aT, wsb + OFF_W3, b3, hb, hF, lane);
  xb[lane] = 0.0f;                      // x0 = 0
  __syncthreads();

  float* ghF = (float*)aT;              // gh [16x192] f32 = 12288 B (reuses sA)

  for (int t = 0; t < PRED_LEN; ++t) {
    // gh = h @ Whh^T + bhh  (M=16, K=64, N=192)
    gemm_stage<12, 2, 64, 64, 192, false, false, true>(hb, wsb + OFF_WHH, bhh,
                                                       nullptr, ghF, lane);
    __syncthreads();

    // gate math + hidden update (16*64 = 1024 elems, 32 uniform iters)
    for (int e = lane; e < 16 * 64; e += 32) {
      const int row = e >> 6, u = e & 63;
      const float x0 = xb[row * 2 + 0], x1 = xb[row * 2 + 1];
      const float ir  = x0 * Wih[u * 2]         + x1 * Wih[u * 2 + 1]         + bih[u];
      const float izv = x0 * Wih[(64 + u) * 2]  + x1 * Wih[(64 + u) * 2 + 1]  + bih[64 + u];
      const float inn = x0 * Wih[(128 + u) * 2] + x1 * Wih[(128 + u) * 2 + 1] + bih[128 + u];
      const float hr = ghF[row * 192 + u];
      const float hz = ghF[row * 192 + 64 + u];
      const float hn = ghF[row * 192 + 128 + u];
      const float r  = 1.0f / (1.0f + __expf(-(ir + hr)));
      const float zg = 1.0f / (1.0f + __expf(-(izv + hz)));
      const float nn = tanhf(inn + r * hn);
      const float hnew = (1.0f - zg) * nn + zg * hF[e];
      hF[e] = hnew;
      hb[e] = to_bf16(hnew);
    }
    __syncthreads();

    // dx = h @ Wo^T + bo ; x += dx ; emit waypoint. lane -> (row, coord)
    {
      const int row = lane >> 1, c = lane & 1;
      float acc = bo[c];
#pragma unroll 8
      for (int u = 0; u < 64; ++u) acc += hF[row * 64 + u] * Wo[c * 64 + u];
      const float xn = xb[lane] + acc;
      xb[lane] = xn;
      const long long g = (long long)(rowBase + row) * PRED_LEN + t;
      out[g * 2 + c] = xn;
    }
    __syncthreads();
  }
}

extern "C" void kernel_launch(void* const* d_in, const int* in_sizes, int n_in,
                              void* d_out, int out_size, void* d_ws, size_t ws_size,
                              hipStream_t stream) {
  const float* z   = (const float*)d_in[0];
  const float* W1  = (const float*)d_in[1];
  const float* b1  = (const float*)d_in[2];
  const float* W2  = (const float*)d_in[3];
  const float* b2  = (const float*)d_in[4];
  const float* W3  = (const float*)d_in[5];
  const float* b3  = (const float*)d_in[6];
  const float* Wih = (const float*)d_in[7];
  const float* Whh = (const float*)d_in[8];
  const float* bih = (const float*)d_in[9];
  const float* bhh = (const float*)d_in[10];
  const float* Wo  = (const float*)d_in[11];
  const float* bo  = (const float*)d_in[12];
  float* out = (float*)d_out;

  bf16* wsb = (bf16*)d_ws;
  convert_weights_kernel<<<(TOT_W + BLOCK - 1) / BLOCK, BLOCK, 0, stream>>>(
      W1, W2, W3, Whh, wsb);

  const int rows = in_sizes[0] / 361;  // 131072
  gru_traj_kernel<<<rows / ROWS_PER_BLOCK, BLOCK, 0, stream>>>(
      z, wsb, b1, b2, b3, Wih, bih, bhh, Wo, bo, out);
}
